// Attention_5016521802123
// MI455X (gfx1250) — matrix operand order repaired and checked by reading, not compiled, not run on hardware
//
#include <hip/hip_runtime.h>
#include <stdint.h>

#define B_SZ 8
#define T_SZ 2048
#define D_SZ 512
#define TQ 32
#define TK 64
#define NTHREADS 256
#define NKT (T_SZ / TK)

typedef __attribute__((ext_vector_type(16))) __bf16 v16bf;
typedef __attribute__((ext_vector_type(8)))  float  v8f;

union Frag {
  v16bf v;
  uint32_t u[8];
  uint4 q[2];
};

__device__ __forceinline__ uint32_t pack_bf16_2(float lo, float hi) {
  uint32_t a = __float_as_uint(lo);
  uint32_t b = __float_as_uint(hi);
  a = (a + 0x7FFFu + ((a >> 16) & 1u)) >> 16;
  b = (b + 0x7FFFu + ((b >> 16) & 1u)) >> 16;
  return (b << 16) | (a & 0xFFFFu);
}
__device__ __forceinline__ uint16_t f2bf(float f) {
  uint32_t a = __float_as_uint(f);
  return (uint16_t)((a + 0x7FFFu + ((a >> 16) & 1u)) >> 16);
}
__device__ __forceinline__ v8f wmma_bf16(const Frag& a, const Frag& b, v8f c) {
  return __builtin_amdgcn_wmma_f32_16x16x32_bf16(false, a.v, false, b.v,
                                                 (short)0, c, false, false);
}

// ---- CDNA5 async global->LDS copy (ASYNCcnt path), with safe fallback ----
typedef int int4v __attribute__((vector_size(16)));
typedef __attribute__((address_space(1))) int4v* gb128_t;  // global int4*
typedef __attribute__((address_space(3))) int4v* lb128_t;  // LDS int4*

__device__ __forceinline__ void async_ld_b128(const void* g, void* l) {
#if __has_builtin(__builtin_amdgcn_global_load_async_to_lds_b128)
  __builtin_amdgcn_global_load_async_to_lds_b128((gb128_t)g, (lb128_t)l, 0, 0);
#else
  *(uint4*)l = *(const uint4*)g;
#endif
}

template <int N>
__device__ __forceinline__ void wait_async() {
#if __has_builtin(__builtin_amdgcn_s_wait_asynccnt)
  __builtin_amdgcn_s_wait_asynccnt(N);
#else
  asm volatile("s_wait_asynccnt %0" ::"i"(N) : "memory");
#endif
}

// ---------------- kernel 0: f32 -> bf16 weight conversion ----------------
__global__ __launch_bounds__(NTHREADS) void cvt_w_kernel(
    const float* __restrict__ w, uint16_t* __restrict__ o, int n) {
  int i = (blockIdx.x * NTHREADS + threadIdx.x) * 2;
  if (i < n) {
    float2 f = *(const float2*)(w + i);
    *(uint32_t*)(o + i) = pack_bf16_2(f.x, f.y);
  }
}

// ---------------- kernel 1: projection  out = in @ W^T + b ----------------
__global__ __launch_bounds__(NTHREADS) void proj_kernel(
    const float* __restrict__ in, const uint16_t* __restrict__ wbf,
    const float* __restrict__ bias, uint16_t* __restrict__ outp, int trans) {
  __shared__ __align__(16) uint16_t sIn[TQ][D_SZ];
  const int tid = threadIdx.x;
  const int lane = tid & 31, wid = tid >> 5;
  const int r0 = blockIdx.x * TQ;
  {
    const float2* src = (const float2*)(in + (size_t)r0 * D_SZ);
    uint32_t* dst = (uint32_t*)&sIn[0][0];
    for (int i = tid; i < TQ * D_SZ / 2; i += NTHREADS) {
      float2 f = src[i];
      dst[i] = pack_bf16_2(f.x, f.y);
    }
  }
  __syncthreads();

  const int rs = wid & 1;
  const int ec = wid >> 1;
  const int ebase = ec * 128;
  const int lhi = (lane & 16) ? 1 : 0;
  const int l15 = lane & 15;
  const int rl = lhi * 8;

  v8f acc[8] = {};
  const int arow = rs * 16 + l15;
  for (int ds = 0; ds < D_SZ / 32; ++ds) {
    Frag a;
    const uint4* ap = (const uint4*)&sIn[arow][ds * 32 + lhi * 16];
    a.q[0] = ap[0];
    a.q[1] = ap[1];
#pragma unroll
    for (int et = 0; et < 8; ++et) {
      const int e = ebase + et * 16 + l15;
      Frag bm;
      const uint4* bp = (const uint4*)&wbf[(size_t)e * D_SZ + ds * 32 + lhi * 16];
      bm.q[0] = bp[0];
      bm.q[1] = bp[1];
      acc[et] = wmma_bf16(a, bm, acc[et]);
    }
  }

  if (!trans) {
#pragma unroll
    for (int et = 0; et < 8; ++et) {
      const int e = ebase + et * 16 + l15;
      const float bv = bias[e];
      for (int r = 0; r < 8; ++r)
        outp[(size_t)(r0 + rs * 16 + rl + r) * D_SZ + e] = f2bf(acc[et][r] + bv);
    }
  } else {
    const int b = r0 / T_SZ;
    const int t0 = (r0 % T_SZ) + rs * 16 + rl;
#pragma unroll
    for (int et = 0; et < 8; ++et) {
      const int e = ebase + et * 16 + l15;
      const float bv = bias[e];
      uint32_t* orow =
          (uint32_t*)(outp + (size_t)b * D_SZ * T_SZ + (size_t)e * T_SZ + t0);
      for (int r = 0; r < 8; r += 2)
        orow[r >> 1] = pack_bf16_2(acc[et][r] + bv, acc[et][r + 1] + bv);
    }
  }
}

// ---------------- kernel 2: fused flash attention ----------------
__global__ __launch_bounds__(NTHREADS) void attn_kernel(
    const uint16_t* __restrict__ qp, const uint16_t* __restrict__ kp,
    const uint16_t* __restrict__ vpT, const float* __restrict__ mask,
    float* __restrict__ out) {
  __shared__ __align__(16) uint16_t sQ[TQ][D_SZ];   // 32 KB (async-filled)
  __shared__ float sS[TQ][TK + 1];                  // scaled scores (f32)
  __shared__ __align__(16) uint16_t sP[TQ][TK];     // probs (bf16)
  __shared__ __align__(16) float sMask[2][TQ][TK];  // double-buffered mask tiles
  __shared__ float sM[TQ], sL[TQ], sC[TQ];
  __shared__ float sRedM[TQ][8], sRedS[TQ][8];

  const int tid = threadIdx.x;
  const int lane = tid & 31, wid = tid >> 5;
  const int b = blockIdx.y;
  const int q0 = blockIdx.x * TQ;
  const int lhi = (lane & 16) ? 1 : 0;
  const int l15 = lane & 15;
  const int rl = lhi * 8;

  // mask tile copy mapping: 32 rows x 64 f32 = 512 x b128 chunks, 2 per thread
  const int mrowi = tid >> 3;
  const int mc0 = tid & 7;
  const float* mtile0 = mask + ((size_t)b * T_SZ + q0) * T_SZ;  // + k0 + row*T

  // ---- prologue: async-fill Q tile and mask tile 0 ----
  {
    const uint16_t* qsrc = qp + ((size_t)b * T_SZ + q0) * D_SZ;
    uint16_t* qdst = &sQ[0][0];
#pragma unroll
    for (int i = 0; i < 8; ++i) {
      const int idx = tid + i * NTHREADS;  // 16B chunk index
      async_ld_b128(qsrc + idx * 8, qdst + idx * 8);
    }
#pragma unroll
    for (int it = 0; it < 2; ++it) {
      const int c = mc0 + it * 8;
      async_ld_b128(mtile0 + (size_t)mrowi * T_SZ + c * 4, &sMask[0][mrowi][c * 4]);
    }
  }
  if (tid < TQ) { sM[tid] = -3.0e38f; sL[tid] = 0.0f; }
  wait_async<0>();
  __syncthreads();

  const int qsA = wid >> 2, ksA = wid & 3;   // phase A: 2 q-sub x 4 k-sub
  const int qsB = wid & 1, ecB = wid >> 1;   // phase B: 2 q-sub x 4 e-chunks
  const int ebase = ecB * 128;

  v8f acc[8] = {};
  const float scale = 0.044194173824159216f;  // 1/sqrt(512)
  const int row = tid >> 3, sub = tid & 7;

  for (int kt = 0; kt < NKT; ++kt) {
    const int k0 = kt * TK;
    const int buf = kt & 1;

    // issue async prefetch of next mask tile into the other buffer
    if (kt + 1 < NKT) {
#pragma unroll
      for (int it = 0; it < 2; ++it) {
        const int c = mc0 + it * 8;
        async_ld_b128(mtile0 + (size_t)mrowi * T_SZ + (kt + 1) * TK + c * 4,
                      &sMask[buf ^ 1][mrowi][c * 4]);
      }
    }

    // ---------- phase A: S = (Q K^T) * scale ----------
    {
      v8f s = {};
      const int arow = qsA * 16 + l15;
      const int kcol = k0 + ksA * 16 + l15;
      const uint16_t* krow = kp + ((size_t)b * T_SZ + kcol) * D_SZ;
      for (int es = 0; es < D_SZ / 32; ++es) {
        Frag a;
        const uint4* ap = (const uint4*)&sQ[arow][es * 32 + lhi * 16];
        a.q[0] = ap[0];
        a.q[1] = ap[1];
        Frag bm;
        const uint4* bp = (const uint4*)(krow + es * 32 + lhi * 16);
        bm.q[0] = bp[0];
        bm.q[1] = bp[1];
        s = wmma_bf16(a, bm, s);
      }
#pragma unroll
      for (int r = 0; r < 8; ++r)
        sS[qsA * 16 + rl + r][ksA * 16 + l15] = s[r] * scale;
    }
    // current mask tile must be resident (own-wave portion) before the barrier
    if (kt + 1 < NKT) wait_async<2>(); else wait_async<0>();
    __syncthreads();

    // ---------- online softmax (mask applied pre-softmax, per reference) ----
    float vals[8];
    float mx = -3.0e38f;
#pragma unroll
    for (int j = 0; j < 8; ++j) {
      vals[j] = sS[row][sub * 8 + j] * sMask[buf][row][sub * 8 + j];
      mx = fmaxf(mx, vals[j]);
    }
    sRedM[row][sub] = mx;
    __syncthreads();
    float rowmax = sRedM[row][0];
#pragma unroll
    for (int j = 1; j < 8; ++j) rowmax = fmaxf(rowmax, sRedM[row][j]);
    const float mold = sM[row];
    const float mnew = fmaxf(mold, rowmax);
    const float corr = __expf(mold - mnew);
    float psum = 0.0f;
#pragma unroll
    for (int j = 0; j < 8; j += 2) {
      float e0 = __expf(vals[j] - mnew);
      float e1 = __expf(vals[j + 1] - mnew);
      psum += e0 + e1;
      *(uint32_t*)&sP[row][sub * 8 + j] = pack_bf16_2(e0, e1);
    }
    sRedS[row][sub] = psum;
    __syncthreads();
    if (sub == 0) {
      float ts = 0.0f;
#pragma unroll
      for (int j = 0; j < 8; ++j) ts += sRedS[row][j];
      sL[row] = sL[row] * corr + ts;
      sM[row] = mnew;
      sC[row] = corr;
    }
    __syncthreads();

    // ---------- phase B: O = O*corr + P V ----------
    {
      const int rb = qsB * 16 + rl;
      float cr[8];
#pragma unroll
      for (int r = 0; r < 8; ++r) cr[r] = sC[rb + r];
#pragma unroll
      for (int et = 0; et < 8; ++et)
#pragma unroll
        for (int r = 0; r < 8; ++r) acc[et][r] *= cr[r];

      const int prow = qsB * 16 + l15;
      for (int ks2 = 0; ks2 < 2; ++ks2) {
        Frag a;
        const uint4* ap = (const uint4*)&sP[prow][ks2 * 32 + lhi * 16];
        a.q[0] = ap[0];
        a.q[1] = ap[1];
        const int krow = k0 + ks2 * 32 + lhi * 16;
        const uint16_t* vbase = vpT + (size_t)b * D_SZ * T_SZ + (size_t)krow;
#pragma unroll
        for (int et = 0; et < 8; ++et) {
          const int e = ebase + et * 16 + l15;
          Frag bm;
          const uint4* bp = (const uint4*)(vbase + (size_t)e * T_SZ);
          bm.q[0] = bp[0];
          bm.q[1] = bp[1];
          acc[et] = wmma_bf16(a, bm, acc[et]);
        }
      }
    }
    // next iteration's post-phase-A barrier protects sP/sC/sMask reuse
  }

  // ---------- epilogue: normalize and store ----------
  const int rb = qsB * 16 + rl;
  float linv[8];
#pragma unroll
  for (int r = 0; r < 8; ++r) linv[r] = 1.0f / sL[rb + r];
#pragma unroll
  for (int et = 0; et < 8; ++et) {
    const int e = ebase + et * 16 + l15;
    float* orow = out + ((size_t)b * T_SZ + q0 + rb) * D_SZ + e;
#pragma unroll
    for (int r = 0; r < 8; ++r) orow[(size_t)r * D_SZ] = acc[et][r] * linv[r];
  }
}

// ---------------- host launcher ----------------
extern "C" void kernel_launch(void* const* d_in, const int* in_sizes, int n_in,
                              void* d_out, int out_size, void* d_ws, size_t ws_size,
                              hipStream_t stream) {
  (void)in_sizes; (void)n_in; (void)out_size; (void)ws_size;
  const float* q  = (const float*)d_in[0];
  const float* k  = (const float*)d_in[1];
  const float* v  = (const float*)d_in[2];
  const float* Wq = (const float*)d_in[3];
  const float* bq = (const float*)d_in[4];
  const float* Wk = (const float*)d_in[5];
  const float* bk = (const float*)d_in[6];
  const float* Wv = (const float*)d_in[7];
  const float* bv = (const float*)d_in[8];
  const float* dm = (const float*)d_in[9];
  float* out = (float*)d_out;

  char* ws = (char*)d_ws;
  uint16_t* wqb = (uint16_t*)(ws);
  uint16_t* wkb = (uint16_t*)(ws + (size_t)512 * 1024);
  uint16_t* wvb = (uint16_t*)(ws + (size_t)1024 * 1024);
  uint16_t* qpb = (uint16_t*)(ws + (size_t)2 * 1024 * 1024);
  uint16_t* kpb = (uint16_t*)(ws + (size_t)(2 + 16) * 1024 * 1024);
  uint16_t* vtb = (uint16_t*)(ws + (size_t)(2 + 32) * 1024 * 1024);

  const int wn = D_SZ * D_SZ;
  cvt_w_kernel<<<dim3(wn / (2 * NTHREADS)), dim3(NTHREADS), 0, stream>>>(Wq, wqb, wn);
  cvt_w_kernel<<<dim3(wn / (2 * NTHREADS)), dim3(NTHREADS), 0, stream>>>(Wk, wkb, wn);
  cvt_w_kernel<<<dim3(wn / (2 * NTHREADS)), dim3(NTHREADS), 0, stream>>>(Wv, wvb, wn);

  dim3 pGrid(B_SZ * T_SZ / TQ);
  proj_kernel<<<pGrid, dim3(NTHREADS), 0, stream>>>(q, wqb, bq, qpb, 0);
  proj_kernel<<<pGrid, dim3(NTHREADS), 0, stream>>>(k, wkb, bk, kpb, 0);
  proj_kernel<<<pGrid, dim3(NTHREADS), 0, stream>>>(v, wvb, bv, vtb, 1);

  dim3 aGrid(T_SZ / TQ, B_SZ);
  attn_kernel<<<aGrid, dim3(NTHREADS), 0, stream>>>(qpb, kpb, vtb, dm, out);
}
